// BilinearInteraction_38611755991218
// MI455X (gfx1250) — matrix operand order repaired and checked by reading, not compiled, not run on hardware
//
#include <hip/hip_runtime.h>

typedef __attribute__((ext_vector_type(16))) __bf16 v16bf;
typedef __attribute__((ext_vector_type(8)))  float  v8f;
typedef __attribute__((ext_vector_type(4)))  float  vf4;
typedef __attribute__((ext_vector_type(4)))  unsigned int vu4;

// 8-arg form: (neg_a, A, neg_b, B, c_mod, C, reuse_a, reuse_b)
#define WMMA_BF16(A, B, C) \
  __builtin_amdgcn_wmma_f32_16x16x32_bf16(false, (A), false, (B), (short)0, (C), false, false)

namespace {
constexpr int kF   = 40;
constexpr int kD   = 64;
constexpr int kP   = kF * (kF - 1) / 2;  // 780 pairs
constexpr int kStr = 72;                 // padded LDS row stride (bf16 elems); 36 dwords -> conflict-free b128
}

__global__ __launch_bounds__(256)
void bilinear_pair_wmma_kernel(const float* __restrict__ x,
                               const float* __restrict__ W,
                               float* __restrict__ out) {
  // W_p transposed (e-major) in LDS, split into bf16 hi / lo planes: 2 * 64*72*2B = 18 KB
  __shared__ __bf16 sW[2][kD * kStr];

  const int p = blockIdx.x;
  // row-major triu_indices(F, k=1): walk rows of shrinking length
  int i = 0, rem = p;
  while (rem >= kF - 1 - i) { rem -= kF - 1 - i; ++i; }
  const int j = i + 1 + rem;

  const int tid = threadIdx.x;

  // ---- Stage W_p: each thread produces 8 consecutive d of one transposed row,
  //      hi/lo bf16 split, stored as one ds_store_b128 per plane.
  //      Global reads stay coalesced (consecutive tid -> consecutive e).
  const float* Wp = W + (size_t)p * kD * kD;
#pragma unroll
  for (int k = 0; k < 2; ++k) {
    const int q  = tid + 256 * k;     // [0, 512): one (e, d-octet) task
    const int e  = q & 63;
    const int d0 = (q >> 6) * 8;      // 0,8,...,56
    union { __bf16 b[8]; vu4 u; } ph, pl;
#pragma unroll
    for (int t = 0; t < 8; ++t) {
      const float w = Wp[(d0 + t) * kD + e];
      const __bf16 h = (__bf16)w;
      ph.b[t] = h;
      pl.b[t] = (__bf16)(w - (float)h);  // bf16x3 residual
    }
    *(vu4*)(&sW[0][e * kStr + d0]) = ph.u;   // 16B-aligned (kStr*2=144, d0*2 mult of 16)
    *(vu4*)(&sW[1][e * kStr + d0]) = pl.u;
  }
  __syncthreads();

  const int lane = tid & 31;
  const int wave = tid >> 5;
  const int l16  = lane & 15;
  const int lhi  = lane >> 4;                       // 0: lanes 0-15, 1: lanes 16-31
  const long batch0 = (long)blockIdx.y * 128 + (long)wave * 16;

  // ---- A operands: Xi rows (M = batch row = l16), contiguous K layout:
  //      lane<16: K = c0..c0+15 with c0=0 ; lane>=16: c0=16
  const float* xi = x + (batch0 + l16) * (kF * kD) + (size_t)i * kD;
  const int c0 = lhi * 16;
  v16bf ah0, al0, ah1, al1;
#pragma unroll
  for (int ks = 0; ks < 2; ++ks) {                  // K-step 0: d 0..31, 1: d 32..63
    const int kb = ks * 32;
    const vf4 f0 = *(const vf4*)(xi + kb + c0 + 0);
    const vf4 f1 = *(const vf4*)(xi + kb + c0 + 4);
    const vf4 f2 = *(const vf4*)(xi + kb + c0 + 8);
    const vf4 f3 = *(const vf4*)(xi + kb + c0 + 12);
    const float fa[16] = {f0[0],f0[1],f0[2],f0[3], f1[0],f1[1],f1[2],f1[3],
                          f2[0],f2[1],f2[2],f2[3], f3[0],f3[1],f3[2],f3[3]};
    v16bf h, l;
#pragma unroll
    for (int k2 = 0; k2 < 16; ++k2) {
      const __bf16 hh = (__bf16)fa[k2];
      h[k2] = hh;
      l[k2] = (__bf16)(fa[k2] - (float)hh);         // bf16x3 residual
    }
    if (ks == 0) { ah0 = h; al0 = l; } else { ah1 = h; al1 = l; }
  }

  // ---- B operand fetch from LDS, contiguous K layout (same as A):
  //      lane n<16 holds col N=n, K=kbase..kbase+15 contiguous; lane n+16 holds K=kbase+16..31
  auto loadB = [&](int half, int kbase, int nt) -> v16bf {
    const int e  = nt * 16 + l16;
    const int dd = kbase + lhi * 16;
    union { vu4 q[2]; v16bf v; } u;
    const vu4* pp = (const vu4*)(&sW[half][e * kStr + dd]);  // 16B aligned
    u.q[0] = pp[0];
    u.q[1] = pp[1];
    return u.v;
  };

  float partial[8] = {0, 0, 0, 0, 0, 0, 0, 0};
  // Xj element for C/D slot (VGPR r, this lane): row m = r + 8*lhi, col e = nt*16 + l16
  const float* xjb = x + (batch0 + lhi * 8) * (kF * kD) + (size_t)j * kD + l16;

#pragma unroll
  for (int nt = 0; nt < 4; ++nt) {                  // 4 N-tiles of the 16x64 product
    const v16bf bh0 = loadB(0, 0, nt);
    const v16bf bl0 = loadB(1, 0, nt);
    const v16bf bh1 = loadB(0, 32, nt);
    const v16bf bl1 = loadB(1, 32, nt);
    v8f acc = {0.f, 0.f, 0.f, 0.f, 0.f, 0.f, 0.f, 0.f};
    // bf16x3: hi*hi + hi*lo + lo*hi, fp32 accumulate, per K-step
    acc = WMMA_BF16(ah0, bh0, acc);
    acc = WMMA_BF16(ah0, bl0, acc);
    acc = WMMA_BF16(al0, bh0, acc);
    acc = WMMA_BF16(ah1, bh1, acc);
    acc = WMMA_BF16(ah1, bl1, acc);
    acc = WMMA_BF16(al1, bh1, acc);
#pragma unroll
    for (int r = 0; r < 8; ++r) {
      const float xv = xjb[(size_t)r * (kF * kD) + nt * 16];
      partial[r] += acc[r] * xv;
    }
  }

  // ---- Row-sum: reduce each row's 16 column-lanes (groups of 16 within the wave)
#pragma unroll
  for (int r = 0; r < 8; ++r) {
    float v = partial[r];
    v += __shfl_xor(v, 1, 16);
    v += __shfl_xor(v, 2, 16);
    v += __shfl_xor(v, 4, 16);
    v += __shfl_xor(v, 8, 16);
    partial[r] = v;
  }
  if (l16 < 8) {
    const long m = batch0 + lhi * 8 + l16;
    out[m * kP + p] = partial[l16];
  }
}

extern "C" void kernel_launch(void* const* d_in, const int* in_sizes, int n_in,
                              void* d_out, int out_size, void* d_ws, size_t ws_size,
                              hipStream_t stream) {
  const float* x = (const float*)d_in[0];   // (B, F, D) fp32
  const float* W = (const float*)d_in[1];   // (P, D, D) fp32
  float* out = (float*)d_out;               // (B, P) fp32
  const int B = in_sizes[0] / (kF * kD);    // 2048
  dim3 grid(kP, B / 128);                   // one WG per (pair, 128-batch slab)
  bilinear_pair_wmma_kernel<<<grid, dim3(256), 0, stream>>>(x, W, out);
  (void)n_in; (void)out_size; (void)d_ws; (void)ws_size;
}
